// SparseFCNet_31825707663907
// MI455X (gfx1250) — matrix operand order repaired and checked by reading, not compiled, not run on hardware
//
#include <hip/hip_runtime.h>

typedef __attribute__((ext_vector_type(16))) _Float16 v16h;
typedef __attribute__((ext_vector_type(8)))  _Float16 v8h;
typedef __attribute__((ext_vector_type(4)))  _Float16 v4h;
typedef __attribute__((ext_vector_type(4)))  float    v4f;
typedef __attribute__((ext_vector_type(8)))  float    v8f;

constexpr int ROWS  = 32;        // batch rows per workgroup
constexpr int ROWT  = ROWS / 16; // 16-row WMMA row-tiles
constexpr int K1PAD = 800;       // 784 padded to a multiple of 32
constexpr int SMEM_BYTES = ROWS * K1PAD * 2   /* sX  f16 */
                         + ROWS * 256 * 4     /* sRaw f32 */
                         + ROWS * 256 * 2;    /* sAct f16 */

// f16 packed-weight offsets inside d_ws (in halfs)
constexpr size_t W1_OFF = 0;                       // 256 x 800
constexpr size_t W2_OFF = W1_OFF + 256 * 800;      // 128 x 256
constexpr size_t W3_OFF = W2_OFF + 128 * 256;      // 64 x 128
constexpr size_t W4_OFF = W3_OFF + 64 * 128;       // 32 x 64
constexpr size_t W5_OFF = W4_OFF + 32 * 64;        // 16 x 32 (N padded 10->16)

// ---- one-shot weight prepack: f32 (Nreal x Kreal) -> f16 (Npad x Kpad), zero padded ----
extern "C" __global__ void prepack_w(const float* __restrict__ W, _Float16* __restrict__ out,
                                     int Nreal, int Kreal, int Kpad, int total) {
    const int idx = blockIdx.x * blockDim.x + threadIdx.x;
    if (idx >= total) return;
    const int n = idx / Kpad;
    const int k = idx - n * Kpad;
    const float v = (n < Nreal && k < Kreal) ? W[(size_t)n * Kreal + k] : 0.f;
    out[idx] = (_Float16)v;
}

// ---- A tile (16x32 f16) from LDS, contiguous K order (matches B) ----
static __device__ __forceinline__ v16h lds_load_a(const _Float16* base, int strideH,
                                                  int kc, int lane) {
    const int m  = lane & 15;
    const int kh = (lane >> 4) << 4;             // 0 or 16
    const _Float16* p = base + m * strideH + kc + kh;
    v8h lo = *(const v8h*)(p);                   // K = kc+kh   .. +7
    v8h hi = *(const v8h*)(p + 8);               // K = kc+kh+8 .. +15
    return __builtin_shufflevector(lo, hi, 0,1,2,3,4,5,6,7,8,9,10,11,12,13,14,15);
}

// ---- B tile (32x16 f16) from packed f16 weights: one unconditional 32B load ----
// lane<16: N=ncol+lane, K = kc..kc+15 ; lane>=16: N=ncol+lane-16, K = kc+16..kc+31
static __device__ __forceinline__ v16h ldg_b(const _Float16* __restrict__ Wp, int Kpad,
                                             int ncol, int kc, int lane) {
    const int n  = ncol + (lane & 15);
    const int k0 = kc + ((lane >> 4) << 4);
    return *(const v16h*)(Wp + (size_t)n * Kpad + k0);   // 32B aligned by construction
}

// ---- one fused GEMM layer: sOut(ROWS x N, f32) = A(ROWS x K) * Wp(N x K)^T + bias ----
static __device__ __forceinline__ void gemm_layer(const _Float16* sA, int strideH, int Kchunks,
                                                  const _Float16* __restrict__ Wp,
                                                  const float* __restrict__ bias,
                                                  int Kpad, int N,
                                                  float* sOut, int wave, int lane) {
    const int ntiles = N >> 4;
    for (int t = wave; t < ntiles; t += 8) {     // wave-uniform: EXEC stays all-ones
        v8f acc[ROWT] = {};
        const int ncol = t << 4;
        for (int kc = 0; kc < Kchunks; ++kc) {
            const v16h b = ldg_b(Wp, Kpad, ncol, kc << 5, lane);
            #pragma unroll
            for (int rt = 0; rt < ROWT; ++rt) {  // reuse B across row-tiles
                const v16h a = lds_load_a(sA + rt * 16 * strideH, strideH, kc << 5, lane);
                acc[rt] = __builtin_amdgcn_wmma_f32_16x16x32_f16(
                    false, a, false, b, (short)0, acc[rt], false, false);
            }
        }
        const int   col   = ncol + (lane & 15);
        const float bv    = bias[col];
        const int   mbase = (lane >> 4) << 3;    // C/D layout: vgpr r -> row mbase+r
        #pragma unroll
        for (int rt = 0; rt < ROWT; ++rt)
            #pragma unroll
            for (int r = 0; r < 8; ++r)
                sOut[(rt * 16 + mbase + r) * N + col] = acc[rt][r] + bv;
    }
}

// ---- exact k-winners: radix search for the k-th largest boosted value ----
// Counting uses wave32 ballot + popcount: scalar threshold, zero cross-lane LDS traffic.
template <int N, int KWIN>
static __device__ __forceinline__ void kwinners_layer(const float* sRaw,
                                                      const float* __restrict__ duty,
                                                      float pct, _Float16* sAct,
                                                      int wave, int lane) {
    constexpr int NV = N / 32;                   // values per lane (wave32)
    float bf[NV];                                // boost factors: per-neuron, row-invariant
    #pragma unroll
    for (int j = 0; j < NV; ++j)
        bf[j] = __expf(1.4f * (pct - duty[lane + (j << 5)]));

    for (int m = wave; m < ROWS; m += 8) {
        float vals[NV]; unsigned ords[NV];
        #pragma unroll
        for (int j = 0; j < NV; ++j) {
            const float v = sRaw[m * N + lane + (j << 5)];
            unsigned u = __float_as_uint(v * bf[j]);
            ords[j] = (u & 0x80000000u) ? ~u : (u | 0x80000000u);  // order-preserving map
            vals[j] = v;
        }
        unsigned thr = 0u;                       // lives in an SGPR
        for (int b = 31; b >= 0; --b) {
            const unsigned cand = thr | (1u << b);
            unsigned c = 0;
            #pragma unroll
            for (int j = 0; j < NV; ++j)
                c += (unsigned)__builtin_popcount(
                        __builtin_amdgcn_ballot_w32(ords[j] >= cand));
            if (c >= KWIN) thr = cand;           // scalar compare + branch
        }
        #pragma unroll
        for (int j = 0; j < NV; ++j) {
            const int n = lane + (j << 5);
            sAct[m * N + n] = (ords[j] >= thr) ? (_Float16)vals[j] : (_Float16)0.0f;
        }
    }
}

extern "C" __global__ __launch_bounds__(256)
void sparse_kwinner_mlp(const float* __restrict__ x,
                        const _Float16* __restrict__ w1p, const float* __restrict__ b1,
                        const _Float16* __restrict__ w2p, const float* __restrict__ b2,
                        const _Float16* __restrict__ w3p, const float* __restrict__ b3,
                        const _Float16* __restrict__ w4p, const float* __restrict__ b4,
                        const _Float16* __restrict__ w5p, const float* __restrict__ b5,
                        const float* __restrict__ d1, const float* __restrict__ d2,
                        const float* __restrict__ d3, const float* __restrict__ d4,
                        float* __restrict__ out) {
    extern __shared__ char smem[];
    _Float16* sX   = (_Float16*)smem;                                       // ROWS x 800 f16
    float*    sRaw = (float*)(smem + ROWS * K1PAD * 2);                     // ROWS x 256 f32
    _Float16* sAct = (_Float16*)(smem + ROWS * K1PAD * 2 + ROWS * 256 * 4); // ROWS x 256 f16

    const int  tid  = threadIdx.x;
    const int  wave = tid >> 5;
    const int  lane = tid & 31;
    const long row0 = (long)blockIdx.x * ROWS;

    // stage 0: x tile -> LDS f16 (branch-free; 784 = 196 float4 per row, never row-crossing)
    const v4f* __restrict__ x4 = (const v4f*)(x + row0 * 784);
    for (int idx = tid; idx < ROWS * 196; idx += 256) {
        const v4f f = __builtin_nontemporal_load(x4 + idx);   // streamed once: keep out of L2
        const int m = idx / 196;
        const int c = (idx - m * 196) << 2;
        v4h h;
        h[0] = (_Float16)f[0]; h[1] = (_Float16)f[1];
        h[2] = (_Float16)f[2]; h[3] = (_Float16)f[3];
        *(v4h*)(sX + m * K1PAD + c) = h;                      // 8B-aligned ds_store_b64
    }
    for (int idx = tid; idx < ROWS * 16; idx += 256) {        // zero pad columns 784..799
        const int m = idx >> 4;
        const int c = idx & 15;
        sX[m * K1PAD + 784 + c] = (_Float16)0.f;
    }
    __syncthreads();

    gemm_layer(sX, K1PAD, 25, w1p, b1, K1PAD, 256, sRaw, wave, lane);
    __syncthreads();
    kwinners_layer<256, 128>(sRaw, d1, 0.5f, sAct, wave, lane);
    __syncthreads();
    gemm_layer(sAct, 256, 8, w2p, b2, 256, 128, sRaw, wave, lane);
    __syncthreads();
    kwinners_layer<128, 13>(sRaw, d2, 0.1f, sAct, wave, lane);
    __syncthreads();
    gemm_layer(sAct, 128, 4, w3p, b3, 128, 64, sRaw, wave, lane);
    __syncthreads();
    kwinners_layer<64, 32>(sRaw, d3, 0.5f, sAct, wave, lane);
    __syncthreads();
    gemm_layer(sAct, 64, 2, w4p, b4, 64, 32, sRaw, wave, lane);
    __syncthreads();
    kwinners_layer<32, 16>(sRaw, d4, 0.5f, sAct, wave, lane);
    __syncthreads();

    // layer 5: N=10 padded to one 16-wide tile per row-tile, store straight to HBM
    if (wave < ROWT) {                            // wave-uniform branch, EXEC all-ones
        v8f acc = {};
        const v16h a = lds_load_a(sAct + wave * 16 * 32, 32, 0, lane);
        const v16h b = ldg_b(w5p, 32, 0, 0, lane);            // rows 10..15 are zero-padded
        acc = __builtin_amdgcn_wmma_f32_16x16x32_f16(false, a, false, b, (short)0, acc,
                                                     false, false);
        const int col = lane & 15;
        if (col < 10) {
            const float bv    = b5[col];
            const int   mbase = (lane >> 4) << 3;
            #pragma unroll
            for (int r = 0; r < 8; ++r)
                out[(row0 + wave * 16 + mbase + r) * 10 + col] = acc[r] + bv;
        }
    }
}

extern "C" void kernel_launch(void* const* d_in, const int* in_sizes, int n_in,
                              void* d_out, int out_size, void* d_ws, size_t ws_size,
                              hipStream_t stream) {
    const float* x  = (const float*)d_in[0];
    const float* w1 = (const float*)d_in[1];
    const float* b1 = (const float*)d_in[2];
    const float* w2 = (const float*)d_in[3];
    const float* b2 = (const float*)d_in[4];
    const float* w3 = (const float*)d_in[5];
    const float* b3 = (const float*)d_in[6];
    const float* w4 = (const float*)d_in[7];
    const float* b4 = (const float*)d_in[8];
    const float* w5 = (const float*)d_in[9];
    const float* b5 = (const float*)d_in[10];
    const float* d1 = (const float*)d_in[11];
    const float* d2 = (const float*)d_in[12];
    const float* d3 = (const float*)d_in[13];
    const float* d4 = (const float*)d_in[14];
    float* out = (float*)d_out;

    _Float16* wsp = (_Float16*)d_ws;
    _Float16* w1p = wsp + W1_OFF;
    _Float16* w2p = wsp + W2_OFF;
    _Float16* w3p = wsp + W3_OFF;
    _Float16* w4p = wsp + W4_OFF;
    _Float16* w5p = wsp + W5_OFF;

    // one-shot f32 -> padded f16 weight prepack (re-done every call; deterministic)
    prepack_w<<<dim3((256 * 800 + 255) / 256), dim3(256), 0, stream>>>(w1, w1p, 256, 784, 800, 256 * 800);
    prepack_w<<<dim3((128 * 256 + 255) / 256), dim3(256), 0, stream>>>(w2, w2p, 128, 256, 256, 128 * 256);
    prepack_w<<<dim3((64 * 128 + 255) / 256),  dim3(256), 0, stream>>>(w3, w3p, 64, 128, 128, 64 * 128);
    prepack_w<<<dim3((32 * 64 + 255) / 256),   dim3(256), 0, stream>>>(w4, w4p, 32, 64, 64, 32 * 64);
    prepack_w<<<dim3((16 * 32 + 255) / 256),   dim3(256), 0, stream>>>(w5, w5p, 10, 32, 32, 16 * 32);

    const int batch  = in_sizes[0] / 784;
    const int blocks = (batch + ROWS - 1) / ROWS;

    (void)hipFuncSetAttribute((const void*)sparse_kwinner_mlp,
                              hipFuncAttributeMaxDynamicSharedMemorySize, SMEM_BYTES);
    sparse_kwinner_mlp<<<dim3(blocks), dim3(256), SMEM_BYTES, stream>>>(
        x, w1p, b1, w2p, b2, w3p, b3, w4p, b4, w5p, b5, d1, d2, d3, d4, out);
}
